// RAdaptGAT_49048526520904
// MI455X (gfx1250) — matrix operand order repaired and checked by reading, not compiled, not run on hardware
//
#include <hip/hip_runtime.h>
#include <hip/hip_bf16.h>
#include <math.h>

// ---------------------------------------------------------------------------
// GATv2 x4 for MI455X (gfx1250, wave32).
// GEMMs: operands pre-packed to f16 WMMA-fragment order in global memory
// (L2-resident), inner loop = global_load_b128 + v_wmma_f32_16x16x32_f16 only.
// Edge phase: float4 L2 gathers + u32-key atomic segment-max + f32 atomics.
// ---------------------------------------------------------------------------

typedef __attribute__((ext_vector_type(16))) _Float16 v16h;
typedef __attribute__((ext_vector_type(8)))  float    v8f;

#define EPSBN 1e-5f
#define NEG_SLOPE 0.2f

// ---------------------------------------------------------------------------
__global__ void zero_f32_kernel(float* __restrict__ p, int n) {
    int i = blockIdx.x * blockDim.x + threadIdx.x;
    int stride = gridDim.x * blockDim.x;
    for (; i < n; i += stride) p[i] = 0.0f;
}

// ---------------------------------------------------------------------------
// Pack activations A[M,128] (f32, row-major) into f16 A-fragments.
// Fragment f = (mtile*4 + ks)*32 + lane holds 16 halves for that lane:
//   row = mtile*16 + (lane&15), akb = (lane<16)?0:8,
//   half j -> A[row][ks*32 + akb + (j<8 ? j : j+8)]
// One thread per fragment: 4x b128 loads, 8x cvt_pk, 2x b128 stores.
// ---------------------------------------------------------------------------
__global__ void pack_a_kernel(const float* __restrict__ A, _Float16* __restrict__ out,
                              int nfrag) {
    int f = blockIdx.x * blockDim.x + threadIdx.x;
    if (f >= nfrag) return;
    int lane  = f & 31;
    int ks    = (f >> 5) & 3;
    int mtile = f >> 7;
    int row   = mtile * 16 + (lane & 15);
    int akb   = (lane < 16) ? 0 : 8;
    const float* p = A + (size_t)row * 128 + ks * 32 + akb;

    float4 lo0 = *(const float4*)(p + 0);
    float4 lo1 = *(const float4*)(p + 4);
    float4 hi0 = *(const float4*)(p + 16);
    float4 hi1 = *(const float4*)(p + 20);

    v16h t;
    t[0] = (_Float16)lo0.x;  t[1] = (_Float16)lo0.y;
    t[2] = (_Float16)lo0.z;  t[3] = (_Float16)lo0.w;
    t[4] = (_Float16)lo1.x;  t[5] = (_Float16)lo1.y;
    t[6] = (_Float16)lo1.z;  t[7] = (_Float16)lo1.w;
    t[8]  = (_Float16)hi0.x; t[9]  = (_Float16)hi0.y;
    t[10] = (_Float16)hi0.z; t[11] = (_Float16)hi0.w;
    t[12] = (_Float16)hi1.x; t[13] = (_Float16)hi1.y;
    t[14] = (_Float16)hi1.z; t[15] = (_Float16)hi1.w;
    *(v16h*)(out + (size_t)f * 16) = t;
}

// ---------------------------------------------------------------------------
// Pack weights W[128,Ncol] (f32, row-major) into f16 B-fragments.
// Fragment f = (ks*NT + nt)*32 + lane:
//   col = nt*16 + (lane&15), bkb = (lane<16)?0:8,
//   half j -> W[ks*32 + bkb + (j<8 ? j : j+8)][col]
// Tiny kernel (<=1024 threads).
// ---------------------------------------------------------------------------
__global__ void pack_b_kernel(const float* __restrict__ W, _Float16* __restrict__ out,
                              int Ncol, int nfrag) {
    int f = blockIdx.x * blockDim.x + threadIdx.x;
    if (f >= nfrag) return;
    int lane = f & 31;
    int nt   = (f >> 5) % (Ncol >> 4);
    int ks   = (f >> 5) / (Ncol >> 4);
    int col  = nt * 16 + (lane & 15);
    int bkb  = (lane < 16) ? 0 : 8;
    const float* p = W + (size_t)(ks * 32 + bkb) * Ncol + col;
    v16h t;
    #pragma unroll
    for (int j = 0; j < 16; ++j) t[j] = (_Float16)p[(size_t)(j < 8 ? j : j + 8) * Ncol];
    *(v16h*)(out + (size_t)f * 16) = t;
}

// ---------------------------------------------------------------------------
// Packed WMMA GEMM: D[M,Ncol] = A * W. One wave -> 16 rows x (NT*16) cols.
// K = 128 fully unrolled (4 k-steps). Inner loop: b128 loads + v_wmma only.
// Grid: (M/16, Ncol/(NT*16)). NTtotal = Ncol/16.
// ---------------------------------------------------------------------------
template <int NT>
__global__ __launch_bounds__(32)
void wmma_gemm_packed(const _Float16* __restrict__ Ah, const _Float16* __restrict__ Bh,
                      float* __restrict__ D, int Ncol, int NTtotal) {
    const int lane  = threadIdx.x;
    const int mtile = blockIdx.x;
    const int nblk  = blockIdx.y;

    v8f acc[NT];
    #pragma unroll
    for (int nt = 0; nt < NT; ++nt) acc[nt] = (v8f){};

    #pragma unroll
    for (int ks = 0; ks < 4; ++ks) {
        v16h a = *(const v16h*)(Ah + ((size_t)((mtile * 4 + ks) * 32 + lane)) * 16);
        #pragma unroll
        for (int nt = 0; nt < NT; ++nt) {
            int ntg = nblk * NT + nt;
            v16h b = *(const v16h*)(Bh + ((size_t)((ks * NTtotal + ntg) * 32 + lane)) * 16);
            acc[nt] = __builtin_amdgcn_wmma_f32_16x16x32_f16(
                false, a, false, b, (short)0, acc[nt], false, false);
        }
    }

    const int col   = lane & 15;
    const int rbase = (lane < 16) ? 0 : 8;
    float* dbase = D + (size_t)(mtile * 16 + rbase) * Ncol + nblk * NT * 16 + col;
    #pragma unroll
    for (int nt = 0; nt < NT; ++nt) {
        #pragma unroll
        for (int r = 0; r < 8; ++r) {
            dbase[(size_t)r * Ncol + nt * 16] = acc[nt][r];
        }
    }
}

// ---------------------------------------------------------------------------
// order-preserving float <-> uint key for atomic segment-max
__device__ __forceinline__ unsigned f2key(float f) {
    unsigned b = __float_as_uint(f);
    return (b & 0x80000000u) ? ~b : (b | 0x80000000u);
}
__device__ __forceinline__ float key2f(unsigned k) {
    unsigned b = (k & 0x80000000u) ? (k & 0x7FFFFFFFu) : ~k;
    return __uint_as_float(b);
}

__device__ __forceinline__ float lrelu(float v) {
    return (v > 0.0f) ? v : (NEG_SLOPE * v);
}

// ---------------------------------------------------------------------------
// Edge pass 1: logit + atomic segment-max per (dst, head)
// ---------------------------------------------------------------------------
__global__ void edge_logit_kernel(const float* __restrict__ xl, const float* __restrict__ xr,
                                  const int* __restrict__ srcI, const int* __restrict__ dstI,
                                  const float* __restrict__ att, float* __restrict__ logits,
                                  unsigned* __restrict__ mxkey,
                                  int E, int ET, int H) {
    int tid = blockIdx.x * blockDim.x + threadIdx.x;
    if (tid >= ET * H) return;
    int e = tid / H;
    int h = tid - e * H;
    int src = (e < E) ? srcI[e] : (e - E);
    int dst = (e < E) ? dstI[e] : (e - E);
    const float4* pl = (const float4*)(xl + ((size_t)src * H + h) * 16);
    const float4* pr = (const float4*)(xr + ((size_t)dst * H + h) * 16);
    const float4* pa = (const float4*)(att + (size_t)h * 16);
    float s = 0.0f;
    #pragma unroll
    for (int q = 0; q < 4; ++q) {
        float4 l = pl[q], r = pr[q], a = pa[q];
        s += lrelu(l.x + r.x) * a.x;
        s += lrelu(l.y + r.y) * a.y;
        s += lrelu(l.z + r.z) * a.z;
        s += lrelu(l.w + r.w) * a.w;
    }
    logits[tid] = s;
    atomicMax(&mxkey[(size_t)dst * H + h], f2key(s));
}

// ---------------------------------------------------------------------------
// Edge pass 2: a = exp(logit - mx); denom += a
// ---------------------------------------------------------------------------
__global__ void edge_exp_kernel(float* __restrict__ aE,
                                const unsigned* __restrict__ mxkey,
                                const int* __restrict__ dstI,
                                float* __restrict__ denom,
                                int E, int ET, int H) {
    int tid = blockIdx.x * blockDim.x + threadIdx.x;
    if (tid >= ET * H) return;
    int e = tid / H;
    int h = tid - e * H;
    int dst = (e < E) ? dstI[e] : (e - E);
    float mx = key2f(mxkey[(size_t)dst * H + h]);
    float a = __expf(aE[tid] - mx);
    aE[tid] = a;
    atomicAdd(&denom[(size_t)dst * H + h], a);
}

// ---------------------------------------------------------------------------
// Edge pass 3: acc[dst][h,:] += xl[src][h,:] * (a/denom)
// ---------------------------------------------------------------------------
__global__ void edge_scatter_kernel(const float* __restrict__ xl,
                                    const int* __restrict__ srcI, const int* __restrict__ dstI,
                                    const float* __restrict__ aE, const float* __restrict__ denom,
                                    float* __restrict__ acc,
                                    int E, int ET, int H) {
    int tid = blockIdx.x * blockDim.x + threadIdx.x;
    if (tid >= ET * H) return;
    int e = tid / H;
    int h = tid - e * H;
    int src = (e < E) ? srcI[e] : (e - E);
    int dst = (e < E) ? dstI[e] : (e - E);
    float alpha = aE[tid] / denom[(size_t)dst * H + h];
    const float4* pl = (const float4*)(xl + ((size_t)src * H + h) * 16);
    float* po = acc + ((size_t)dst * H + h) * 16;
    #pragma unroll
    for (int q = 0; q < 4; ++q) {
        float4 l = pl[q];
        atomicAdd(&po[q * 4 + 0], l.x * alpha);
        atomicAdd(&po[q * 4 + 1], l.y * alpha);
        atomicAdd(&po[q * 4 + 2], l.z * alpha);
        atomicAdd(&po[q * 4 + 3], l.w * alpha);
    }
}

// ---------------------------------------------------------------------------
// Finalize layers 0..2: + bias, BN(eval), ELU (in place)
// ---------------------------------------------------------------------------
__global__ void finalize_bn_elu_kernel(float* __restrict__ acc,
                                       const float* __restrict__ bias,
                                       const float* __restrict__ g, const float* __restrict__ be,
                                       const float* __restrict__ m, const float* __restrict__ var,
                                       int n) {
    int tid = blockIdx.x * blockDim.x + threadIdx.x;
    if (tid >= n) return;
    int j = tid & 127;
    float v = acc[tid] + bias[j];
    v = (v - m[j]) * rsqrtf(var[j] + EPSBN) * g[j] + be[j];
    v = (v > 0.0f) ? v : (__expf(v) - 1.0f);
    acc[tid] = v;
}

__global__ void finalize_bias_kernel(float* __restrict__ acc,
                                     const float* __restrict__ bias, int n) {
    int tid = blockIdx.x * blockDim.x + threadIdx.x;
    if (tid >= n) return;
    acc[tid] = acc[tid] + bias[tid & 15];
}

// ---------------------------------------------------------------------------
extern "C" void kernel_launch(void* const* d_in, const int* in_sizes, int n_in,
                              void* d_out, int out_size, void* d_ws, size_t ws_size,
                              hipStream_t stream) {
    const int IN_C = 128, HC = 128, Hh = 8, OUTC = 16;

    const float* x = (const float*)d_in[0];
    const int*   ei = (const int*)d_in[1];
    const int N = in_sizes[0] / IN_C;           // 50000
    const int E = in_sizes[1] / 2;              // 500000
    const int ET = E + N;                       // with self loops
    const int* srcI = ei;
    const int* dstI = ei + E;

    const float* Wl[4]   = {(const float*)d_in[2],  (const float*)d_in[6],
                            (const float*)d_in[10], (const float*)d_in[14]};
    const float* Wr[4]   = {(const float*)d_in[3],  (const float*)d_in[7],
                            (const float*)d_in[11], (const float*)d_in[15]};
    const float* att[4]  = {(const float*)d_in[4],  (const float*)d_in[8],
                            (const float*)d_in[12], (const float*)d_in[16]};
    const float* bias[4] = {(const float*)d_in[5],  (const float*)d_in[9],
                            (const float*)d_in[13], (const float*)d_in[17]};
    const float* bnG[3]  = {(const float*)d_in[18], (const float*)d_in[22], (const float*)d_in[26]};
    const float* bnB[3]  = {(const float*)d_in[19], (const float*)d_in[23], (const float*)d_in[27]};
    const float* bnM[3]  = {(const float*)d_in[20], (const float*)d_in[24], (const float*)d_in[28]};
    const float* bnV[3]  = {(const float*)d_in[21], (const float*)d_in[25], (const float*)d_in[29]};

    // workspace layout
    float* ws = (float*)d_ws;
    float*    hA    = ws;                                // N*128
    float*    hB    = hA + (size_t)N * HC;               // N*128
    float*    xl    = hB + (size_t)N * HC;               // N*128
    float*    xr    = xl + (size_t)N * HC;               // N*128
    float*    aE    = xr + (size_t)N * HC;               // ET*8 (logits -> alphas)
    unsigned* mxk   = (unsigned*)(aE + (size_t)ET * Hh); // N*8 keys
    float*    denom = (float*)(mxk + (size_t)N * Hh);    // N*8
    _Float16* Ah    = (_Float16*)(denom + (size_t)N * Hh); // N*128 halves (A fragments)
    _Float16* Whl   = Ah + (size_t)N * HC;               // 128*128 halves
    _Float16* Whr   = Whl + (size_t)HC * HC;             // 128*128 halves

    const int ZB = 256;
    const int EB = 256;
    const int MT = N / 16;                 // 3125 m-tiles (N divisible by 16)
    const int nfragA = MT * 4 * 32;        // A fragments

    for (int i = 0; i < 4; ++i) {
        const int Hl   = (i < 3) ? Hh : 1;
        const int Cout = Hl * 16;          // 128 or 16
        const float* hin = (i == 0) ? x : ((i == 1) ? hA : ((i == 2) ? hB : hA));
        float* acc = (i == 0) ? hA : ((i == 1) ? hB : ((i == 2) ? hA : (float*)d_out));

        // zero accumulators / softmax state
        {
            int n0 = N * Cout;
            zero_f32_kernel<<<(n0 + ZB - 1) / ZB, ZB, 0, stream>>>(acc, n0);
            int n1 = N * Hl;
            zero_f32_kernel<<<(n1 + ZB - 1) / ZB, ZB, 0, stream>>>(denom, n1);
            zero_f32_kernel<<<(n1 + ZB - 1) / ZB, ZB, 0, stream>>>((float*)mxk, n1);
        }

        // pack operands to f16 fragment order
        pack_a_kernel<<<(nfragA + ZB - 1) / ZB, ZB, 0, stream>>>(hin, Ah, nfragA);
        {
            int nfragB = 4 * (Cout / 16) * 32;   // 1024 or 128
            pack_b_kernel<<<(nfragB + 127) / 128, 128, 0, stream>>>(Wl[i], Whl, Cout, nfragB);
            pack_b_kernel<<<(nfragB + 127) / 128, 128, 0, stream>>>(Wr[i], Whr, Cout, nfragB);
        }

        // WMMA GEMMs: xl = h@Wl, xr = h@Wr
        if (Cout == 128) {
            dim3 grid(MT, 2);   // 16 x 64 per wave, NTtotal = 8
            wmma_gemm_packed<4><<<grid, 32, 0, stream>>>(Ah, Whl, xl, 128, 8);
            wmma_gemm_packed<4><<<grid, 32, 0, stream>>>(Ah, Whr, xr, 128, 8);
        } else {
            dim3 grid(MT, 1);   // 16 x 16 per wave, NTtotal = 1
            wmma_gemm_packed<1><<<grid, 32, 0, stream>>>(Ah, Whl, xl, 16, 1);
            wmma_gemm_packed<1><<<grid, 32, 0, stream>>>(Ah, Whr, xr, 16, 1);
        }

        // edge phase
        {
            int tE = ET * Hl;
            int blocks = (tE + EB - 1) / EB;
            edge_logit_kernel<<<blocks, EB, 0, stream>>>(xl, xr, srcI, dstI, att[i],
                                                         aE, mxk, E, ET, Hl);
            edge_exp_kernel<<<blocks, EB, 0, stream>>>(aE, mxk, dstI, denom, E, ET, Hl);
            edge_scatter_kernel<<<blocks, EB, 0, stream>>>(xl, srcI, dstI, aE, denom,
                                                           acc, E, ET, Hl);
        }

        // finalize
        if (i < 3) {
            int n0 = N * HC;
            finalize_bn_elu_kernel<<<(n0 + ZB - 1) / ZB, ZB, 0, stream>>>(
                acc, bias[i], bnG[i], bnB[i], bnM[i], bnV[i], n0);
        } else {
            int n0 = N * OUTC;
            finalize_bias_kernel<<<(n0 + ZB - 1) / ZB, ZB, 0, stream>>>(acc, bias[i], n0);
        }
    }
}
